// LGCL_63084479644159
// MI455X (gfx1250) — matrix operand order repaired and checked by reading, not compiled, not run on hardware
//
#include <hip/hip_runtime.h>
#include <hip/hip_bf16.h>

#define N_NODES 10000
#define D_FEAT  128
#define KTOP    8
#define DEG     32
#define OUT_CH  128
#define KK_REAL 81     // (K+1) * IN_CH = 9*9
#define KK_PAD  96     // pad K-dim to 3 WMMA k-steps of 32
#define W_OUT   120    // D - K

typedef __attribute__((ext_vector_type(16))) _Float16 v16h;
typedef __attribute__((ext_vector_type(8)))  _Float16 v8h;
typedef __attribute__((ext_vector_type(8)))  float    v8f;
typedef __attribute__((ext_vector_type(4)))  int      v4i;

#define AS1 __attribute__((address_space(1)))
#define AS3 __attribute__((address_space(3)))

#if defined(__gfx1250__) && \
    __has_builtin(__builtin_amdgcn_global_load_async_to_lds_b128) && \
    __has_builtin(__builtin_amdgcn_s_wait_asynccnt)
#define HAVE_ASYNC_LDS 1
#else
#define HAVE_ASYNC_LDS 0
#endif

__device__ inline v16h cat8(v8h lo, v8h hi) {
  return __builtin_shufflevector(lo, hi, 0,1,2,3,4,5,6,7,8,9,10,11,12,13,14,15);
}

// ---- Kernel 1: weight repack f32 [128][81] -> f16 [128][96] (zero padded) ----
__global__ void prep_w(const float* __restrict__ conv_w, _Float16* __restrict__ w_h) {
  int o = threadIdx.x;                       // 128 threads, one output channel each
  for (int kk = 0; kk < KK_PAD; ++kk) {
    float v = (kk < KK_REAL) ? conv_w[o * KK_REAL + kk] : 0.0f;
    w_h[o * KK_PAD + kk] = (_Float16)v;
  }
}

// ---- Kernel 2: scan one adjacency row per block (b128 stream + prefetch) ----
__global__ __launch_bounds__(256)
void scan_adj(const int* __restrict__ adj, int* __restrict__ nbr) {
  const int n = blockIdx.x;
  __shared__ int cnt;
  if (threadIdx.x == 0) cnt = 0;
  if (threadIdx.x < DEG) nbr[n * DEG + threadIdx.x] = -1;
  __syncthreads();
  // row base = n*40000 B, 16B aligned; 10000 ints = 2500 int4
  const int4* row4 = (const int4*)(adj + (size_t)n * N_NODES);
  for (int c4 = threadIdx.x; c4 < N_NODES / 4; c4 += 256) {
    __builtin_prefetch(row4 + c4 + 2048, 0, 1);   // ~8 strides ahead
    int4 v = row4[c4];
    if (v.x | v.y | v.z | v.w) {
      if (v.x) { int p = atomicAdd(&cnt, 1); if (p < DEG) nbr[n * DEG + p] = 4 * c4 + 0; }
      if (v.y) { int p = atomicAdd(&cnt, 1); if (p < DEG) nbr[n * DEG + p] = 4 * c4 + 1; }
      if (v.z) { int p = atomicAdd(&cnt, 1); if (p < DEG) nbr[n * DEG + p] = 4 * c4 + 2; }
      if (v.w) { int p = atomicAdd(&cnt, 1); if (p < DEG) nbr[n * DEG + p] = 4 * c4 + 3; }
    }
  }
}

// ---- Kernel 3: fused gather + per-feature top-8 + im2col + WMMA GEMM ----
__global__ __launch_bounds__(256, 2)
void lgcl_conv(const float* __restrict__ feat,
               const int*   __restrict__ nbr,
               const _Float16* __restrict__ w_h,
               const float* __restrict__ bias,
               float* __restrict__ out) {
  __shared__ _Float16 lds_sel[KTOP + 1][D_FEAT];   // 9 x 128   (2304 B)
  __shared__ _Float16 lds_Xt[D_FEAT][KK_PAD];      // 128 x 96  (24576 B) im2col, w-major
  __shared__ _Float16 lds_W[OUT_CH][KK_PAD];       // 128 x 96  (24576 B)
  __shared__ int nbr_s[DEG];

  const int n   = blockIdx.x;
  const int tid = threadIdx.x;

  // stage neighbor list + weights (weights are L2-resident, 24.5 KB)
  if (tid < DEG) nbr_s[tid] = nbr[n * DEG + tid];
#if HAVE_ASYNC_LDS
  {
    // 1536 x 16B chunks; each thread issues 6 async b128 copies, ASYNCcnt-tracked
    for (int i = tid; i < (OUT_CH * KK_PAD) / 8; i += 256) {
      __builtin_amdgcn_global_load_async_to_lds_b128(
          (AS1 v4i*)(w_h + (size_t)i * 8),
          (AS3 v4i*)(&lds_W[0][0] + (size_t)i * 8),
          0, 0);
    }
    __builtin_amdgcn_s_wait_asynccnt(0);
  }
#else
  {
    const uint4* src = (const uint4*)w_h;          // 1536 uint4
    uint4* dst = (uint4*)&lds_W[0][0];
    for (int i = tid; i < (OUT_CH * KK_PAD) / 8; i += 256) dst[i] = src[i];
  }
#endif
  __syncthreads();

  // per-feature top-8 over <=32 neighbors (register compare-swap chain)
  if (tid < D_FEAT) {
    const int d = tid;
    lds_sel[0][d] = (_Float16)feat[(size_t)n * D_FEAT + d];
    float top[KTOP];
#pragma unroll
    for (int r = 0; r < KTOP; ++r) top[r] = -3.0e38f;
#pragma unroll
    for (int j = 0; j < DEG; ++j) {
      const int idx = nbr_s[j];
      float v = (idx >= 0) ? feat[(size_t)idx * D_FEAT + d] : -3.0e38f;
#pragma unroll
      for (int r = 0; r < KTOP; ++r) {           // insert v into sorted top[]
        float mx = fmaxf(top[r], v);
        float mn = fminf(top[r], v);
        top[r] = mx; v = mn;
      }
    }
#pragma unroll
    for (int r = 0; r < KTOP; ++r) {
      float v = top[r];
      if (v < -1.0e38f) v = 0.0f;                // zero-pad missing neighbors
      lds_sel[1 + r][d] = (_Float16)v;
    }
  }
  __syncthreads();

  // im2col, transposed: Xt[w][kk] = sel[kk/9][w + kk%9], zero pad w>=120, kk>=81
  for (int e = tid; e < D_FEAT * KK_PAD; e += 256) {
    const int w  = e / KK_PAD;
    const int kk = e - w * KK_PAD;
    _Float16 v = (_Float16)0.0f;
    if (w < W_OUT && kk < KK_REAL) {
      const int i = kk / 9;
      const int t = kk - i * 9;
      v = lds_sel[i][w + t];
    }
    lds_Xt[w][kk] = v;
  }
  __syncthreads();

  // GEMM: out[128 x 120] = W[128 x 96] * Xt^T[96 x 128], 8 waves x 8 n-tiles x 3 k
  const int wave = tid >> 5;
  const int lane = tid & 31;
  const int m0   = wave * 16;
  const int Mr   = lane & 15;      // A: M row within tile / B: N col within tile
  const int hg   = lane >> 4;      // half-group

  // A fragments: reused across all 8 N tiles
  v16h a[3];
#pragma unroll
  for (int ks = 0; ks < 3; ++ks) {
    const int k0 = ks * 32;
    const _Float16* p0 = &lds_W[m0 + Mr][k0 + hg * 8];
    a[ks] = cat8(*(const v8h*)p0, *(const v8h*)(p0 + 16));
  }

  float brow[8];
#pragma unroll
  for (int r = 0; r < 8; ++r) brow[r] = bias[m0 + hg * 8 + r];

#pragma unroll
  for (int j = 0; j < 8; ++j) {
    const int w0 = j * 16;
    v8f acc = {};
#pragma unroll
    for (int ks = 0; ks < 3; ++ks) {
      const int k0 = ks * 32;
      const _Float16* q0 = &lds_Xt[w0 + Mr][k0 + hg * 8];
      v16h b = cat8(*(const v8h*)q0, *(const v8h*)(q0 + 16));
      acc = __builtin_amdgcn_wmma_f32_16x16x32_f16(
          false, a[ks], false, b, (short)0, acc, false, false);
    }
    const int w = w0 + Mr;
    if (w < W_OUT) {
#pragma unroll
      for (int r = 0; r < 8; ++r) {
        const int o = m0 + hg * 8 + r;
        out[((size_t)n * OUT_CH + o) * W_OUT + w] = acc[r] + brow[r];
      }
    }
  }
}

extern "C" void kernel_launch(void* const* d_in, const int* in_sizes, int n_in,
                              void* d_out, int out_size, void* d_ws, size_t ws_size,
                              hipStream_t stream) {
  const float* feat   = (const float*)d_in[0];   // [N, 128] f32
  const int*   adj    = (const int*)d_in[1];     // [N, N] i32
  const float* conv_w = (const float*)d_in[2];   // [128, 9, 9] f32
  const float* conv_b = (const float*)d_in[3];   // [128] f32
  float* out = (float*)d_out;                    // [N, 128, 120] f32

  // workspace layout: [w_h f16 128*96][nbr int N*32]  (~1.3 MB total)
  _Float16* w_h = (_Float16*)d_ws;
  int* nbr = (int*)((char*)d_ws + (size_t)OUT_CH * KK_PAD * sizeof(_Float16));

  prep_w  <<<1,        128, 0, stream>>>(conv_w, w_h);
  scan_adj<<<N_NODES,  256, 0, stream>>>(adj, nbr);
  lgcl_conv<<<N_NODES, 256, 0, stream>>>(feat, nbr, w_h, conv_b, out);
}
